// HFPS_79886391706304
// MI455X (gfx1250) — matrix operand order repaired and checked by reading, not compiled, not run on hardware
//
#include <hip/hip_runtime.h>
#include <hip/hip_bf16.h>
#include <math.h>

#define LSZ   48
#define NPIX  2304          // 48*48
#define MSIZE 4608          // 2*48*48
#define CHN   64            // NT2 + JASTROW
#define NT2c  48
#define NA    1048          // 1024 + 24
#define LDA   1056          // padded leading dim (multiple of 32)

typedef __attribute__((ext_vector_type(16))) __bf16 v16bf;
typedef __attribute__((ext_vector_type(8)))  float  v8f;

// ---------- helpers ----------
__device__ __forceinline__ unsigned short f2bfbits(float f) {
    union { float f; unsigned u; } c; c.f = f;
    unsigned r = c.u + 0x7FFFu + ((c.u >> 16) & 1u);   // round-to-nearest-even
    return (unsigned short)(r >> 16);
}

__device__ __forceinline__ __bf16 f2bf(float f) {
    union { unsigned short s; __bf16 b; } o; o.s = f2bfbits(f);
    return o.b;
}

__device__ __forceinline__ float gelu_f(float x) {
    // jax.nn.gelu default (tanh approximation)
    float x3 = x * x * x;
    float t  = tanhf(0.7978845608028654f * (x + 0.044715f * x3));
    return 0.5f * x * (1.0f + t);
}

// ---------- elementwise pre-activation: t = act(h / sqrt(i+1)) ----------
__global__ __launch_bounds__(256) void preact_kernel(
    const float* __restrict__ in, float* __restrict__ out,
    int count, float scale, int firstBlock)
{
    int i = blockIdx.x * 256 + threadIdx.x;
    if (i >= count) return;
    float v = in[i] * scale;
    v = firstBlock ? (v * 0.7071067811865476f) : gelu_f(v);
    out[i] = v;
}

// ---------- 3x3 circular conv as implicit GEMM via WMMA (bf16 -> f32) -----
// Grid: 144 pixel tiles (16 consecutive pixels of one row each).
// Block: 128 threads = 4 waves; wave w computes out-channels [16w,16w+16).
// GEMM: C[co,pix] = sum_k W[co,k] * patch[k,pix], k = ci*9 + ky*3 + kx.
// B panel is im2col'd to bf16 in fragment order ONCE per WG, so the per-wave
// inner loop is: 2x contiguous global A runs + 1 aligned LDS vector B load +
// one v_wmma per 32-K chunk (fully unrolled for CIN=64: K=576=18*32 exactly).
template<int CIN>
__global__ __launch_bounds__(128) void conv_wmma_kernel(
    const float* __restrict__ in,    // [CIN][2304]
    const float* __restrict__ W,     // [64][CIN*9]
    const float* __restrict__ bias,  // [64] or nullptr
    const float* __restrict__ res,   // residual [resCh][2304] or nullptr
    float* __restrict__ out,         // [64][2304]
    int resRepeat, int applyGelu)
{
    constexpr int K      = CIN * 9;            // 18 or 576
    constexpr int NCHUNK = (K + 31) / 32;      // 1 or 18

    __shared__ float patch[CIN * 54];          // CIN * 3 rows * 18 cols (wrapped)
    __shared__ __align__(32) unsigned short fragB[NCHUNK * 32 * 16]; // bf16 bits

    const int tile = blockIdx.x;               // 0..143
    const int y    = tile / 3;
    const int x0   = (tile % 3) * 16;
    const int tid  = threadIdx.x;
    const int wave = tid >> 5;
    const int lane = tid & 31;

    // phase 1: cooperative load of the circularly-padded input patch
    for (int e = tid; e < CIN * 54; e += 128) {
        int ci  = e / 54;
        int rem = e - ci * 54;
        int ky  = rem / 18;
        int c   = rem - ky * 18;
        int gy  = y + ky - 1;  gy += (gy < 0) ? LSZ : 0;  gy -= (gy >= LSZ) ? LSZ : 0;
        int gx  = x0 + c - 1;  gx += (gx < 0) ? LSZ : 0;  gx -= (gx >= LSZ) ? LSZ : 0;
        patch[e] = in[ci * NPIX + gy * LSZ + gx];
    }
    __syncthreads();

    // phase 2: build bf16 B panel in WMMA fragment order.
    // B-matrix 32x16 contiguous k order (this code's choice): lanes 0-15 hold K=0..15 (elem j = K j),
    // lanes 16-31 hold K=16..31; element (chunk c, lane l, j) <-> k = 32c+16*(l>>4)+j.
    for (int e = tid; e < NCHUNK * 512; e += 128) {
        int c = e >> 9;
        int l = (e >> 4) & 31;
        int j = e & 15;
        int k = c * 32 + ((l >> 4) << 4) + j;
        unsigned short b = 0;
        if (k < K) {
            int ci = k / 9;
            int r9 = k - ci * 9;
            int ky = r9 / 3;
            int kx = r9 - ky * 3;
            b = f2bfbits(patch[ci * 54 + ky * 18 + (l & 15) + kx]);
        }
        fragB[e] = b;
    }
    __syncthreads();

    const int half   = lane >> 4;
    const int idx    = lane & 15;
    const int coBase = wave * 16;
    const float* wrow = W + (coBase + idx) * K;

    v8f acc = {};
#pragma unroll
    for (int kc0 = 0; kc0 < NCHUNK; ++kc0) {
        const int kc = kc0 * 32;
        v16bf a;
#pragma unroll
        for (int j = 0; j < 16; ++j) {
            // A-matrix 16x32 bf16 layout (matches fragB k order):
            // lanes 0-15: K=0..15 (elem j = K j); lanes 16-31: K=16..31
            int ka = kc + 16 * half + j;
            float av;
            if constexpr ((K & 31) == 0) av = wrow[ka];            // no tail
            else av = (ka < K) ? wrow[ka] : 0.0f;
            a[j] = f2bf(av);
        }
        v16bf b = *reinterpret_cast<const v16bf*>(&fragB[(kc0 * 32 + lane) * 16]);
        acc = __builtin_amdgcn_wmma_f32_16x16x32_bf16(
                  false, a, false, b, (short)0, acc, false, false);
    }

    // C/D layout: VGPR r holds row (r + 8*half), column idx
    const int px = x0 + idx;
#pragma unroll
    for (int r = 0; r < 8; ++r) {
        int co = coBase + 8 * half + r;
        float v = acc[r];
        if (bias) v += bias[co];
        if (applyGelu) v = gelu_f(v);
        if (res) v += res[(co / resRepeat) * NPIX + y * LSZ + px];
        out[co * NPIX + y * LSZ + px] = v;
    }
}

// ---------- ordered gather of the 1024 occupied sites (x == +1) ----------
__global__ __launch_bounds__(256) void gather_occ_kernel(
    const float* __restrict__ x, int* __restrict__ nIdx)
{
    __shared__ int cnt[256];
    const int tid = threadIdx.x;
    const int base = tid * 18;                 // 256 * 18 = 4608
    int c = 0;
    for (int i = 0; i < 18; ++i) c += (x[base + i] == 1.0f) ? 1 : 0;
    cnt[tid] = c;
    __syncthreads();
    for (int off = 1; off < 256; off <<= 1) {  // inclusive scan
        int t = (tid >= off) ? cnt[tid - off] : 0;
        __syncthreads();
        cnt[tid] += t;
        __syncthreads();
    }
    int w = cnt[tid] - c;                      // exclusive prefix
    for (int i = 0; i < 18; ++i)
        if (x[base + i] == 1.0f) nIdx[w++] = base + i;
}

// ---------- log_J = sum(out[48:64]) / sqrt(5) ----------
__global__ __launch_bounds__(256) void logj_kernel(
    const float* __restrict__ h, float* __restrict__ outLogJ)
{
    __shared__ float red[256];
    const int tid = threadIdx.x;
    const float* p = h + NT2c * NPIX;
    float s = 0.0f;
    for (int i = tid; i < 16 * NPIX; i += 256) s += p[i];
    red[tid] = s;
    __syncthreads();
    for (int off = 128; off > 0; off >>= 1) {
        if (tid < off) red[tid] += red[tid + off];
        __syncthreads();
    }
    if (tid == 0) *outLogJ = red[0] * 0.4472135954999579f;  // 1/sqrt(5)
}

// ---------- assemble antisymmetric A (1048x1048, lda=1056) ----------
__global__ __launch_bounds__(256) void assemble_A_kernel(
    const float* __restrict__ Fvv, const float* __restrict__ Fhh,
    const float* __restrict__ h, const int* __restrict__ nIdx,
    float* __restrict__ A)
{
    int id = blockIdx.x * 256 + threadIdx.x;
    if (id >= NA * NA) return;
    int i = id / NA, j = id - i * NA;
    const float s5 = 0.4472135954999579f;      // 1/sqrt(5)
    float val;
    if (i < 1024 && j < 1024) {
        int ni = nIdx[i], nj = nIdx[j];
        val = 0.5f * (Fvv[ni * MSIZE + nj] - Fvv[nj * MSIZE + ni]);
    } else if (i < 1024) {
        int b = j - 1024, ni = nIdx[i];
        int ch = 2 * b + ((ni >= NPIX) ? 1 : 0);     // F_vh[m,r] = out[2r + m/2304][m%2304]
        val = h[ch * NPIX + (ni % NPIX)] * s5;
    } else if (j < 1024) {
        int a = i - 1024, nj = nIdx[j];
        int ch = 2 * a + ((nj >= NPIX) ? 1 : 0);
        val = -h[ch * NPIX + (nj % NPIX)] * s5;
    } else {
        int a = i - 1024, b = j - 1024;
        val = 0.5f * (Fhh[a * 24 + b] - Fhh[b * 24 + a]);
    }
    A[i * LDA + j] = val;
}

// ---------- persistent Householder tridiagonalization -> sign & log|Pf| ----
// One 1024-thread workgroup; A (4.4 MB) streams through L2. GEMV and rank-2
// updates are row-per-wave (coalesced), reductions via LDS. Software prefetch
// (global_prefetch_b8) pulls each wave's next row while the current streams.
__global__ __launch_bounds__(1024) void pfaffian_kernel(
    float* __restrict__ A, const float* __restrict__ logJptr,
    float* __restrict__ outp)
{
    __shared__ float v[NA];
    __shared__ float u[NA];
    __shared__ float red[1024];
    const int tid = threadIdx.x;
    const int wv  = tid >> 5;
    const int ln  = tid & 31;
    float ds = 1.0f;

    for (int k = 0; k < NA - 2; ++k) {
        // ||x||, x = column k below row k
        float loc = 0.0f;
        for (int i = tid; i < NA; i += 1024)
            if (i > k) { float c = A[i * LDA + k]; loc += c * c; }
        red[tid] = loc;
        __syncthreads();
        for (int off = 512; off > 0; off >>= 1) {
            if (tid < off) red[tid] += red[tid + off];
            __syncthreads();
        }
        float normx = sqrtf(red[0]);
        float s = (A[(k + 1) * LDA + k] >= 0.0f) ? 1.0f : -1.0f;

        // v = x + s*||x||*e_{k+1}
        for (int i = tid; i < NA; i += 1024) {
            float vi = (i > k) ? A[i * LDA + k] : 0.0f;
            if (i == k + 1) vi += s * normx;
            v[i] = vi;
        }
        __syncthreads();

        loc = 0.0f;
        for (int i = tid; i < NA; i += 1024) loc += v[i] * v[i];
        red[tid] = loc;
        __syncthreads();
        for (int off = 512; off > 0; off >>= 1) {
            if (tid < off) red[tid] += red[tid + off];
            __syncthreads();
        }
        float vn = sqrtf(red[0]);
        __syncthreads();
        bool ok = vn > 1e-30f;
        float inv = ok ? (1.0f / vn) : 0.0f;
        for (int i = tid; i < NA; i += 1024) v[i] *= inv;
        __syncthreads();

        // u = A v  (row per wave, lanes split columns, wave reduce)
        for (int i = wv; i < NA; i += 32) {
            const float* row = A + i * LDA;
            if (i + 32 < NA)
                __builtin_prefetch(row + 32 * LDA + ln * 32, 0, 1);
            float acc = 0.0f;
            for (int j = ln; j < NA; j += 32) acc = fmaf(row[j], v[j], acc);
            for (int off = 16; off > 0; off >>= 1) acc += __shfl_down(acc, off);
            if (ln == 0) u[i] = acc;
        }
        __syncthreads();

        // A += 2 v u^T - 2 u v^T
        for (int i = wv; i < NA; i += 32) {
            float vi2 = 2.0f * v[i], ui2 = 2.0f * u[i];
            float* row = A + i * LDA;
            if (i + 32 < NA)
                __builtin_prefetch(row + 32 * LDA + ln * 32, 0, 1);
            for (int j = ln; j < NA; j += 32)
                row[j] += vi2 * u[j] - ui2 * v[j];
        }
        __syncthreads();

        if (ok) ds = -ds;   // uniform across threads
    }

    // Tridiagonal now: sup = diag(A,1)[::2];
    // sign = ds * prod(sign(sup)); log|Pf| = 0.5*log|det| = sum log|sup|
    float lls = 0.0f; int neg = 0;
    for (int t = tid; t < NA / 2; t += 1024) {
        float a = A[(2 * t) * LDA + (2 * t + 1)];
        neg += (a < 0.0f) ? 1 : 0;
        lls += logf(fabsf(a));
    }
    red[tid] = lls;
    __syncthreads();
    for (int off = 512; off > 0; off >>= 1) {
        if (tid < off) red[tid] += red[tid + off];
        __syncthreads();
    }
    float logsum = red[0];
    __syncthreads();
    red[tid] = (float)neg;
    __syncthreads();
    for (int off = 512; off > 0; off >>= 1) {
        if (tid < off) red[tid] += red[tid + off];
        __syncthreads();
    }
    if (tid == 0) {
        int negTot = (int)red[0];
        outp[0] = ds * ((negTot & 1) ? -1.0f : 1.0f);
        outp[1] = logsum + *logJptr;
    }
}

// ---------- host orchestration ----------
extern "C" void kernel_launch(void* const* d_in, const int* in_sizes, int n_in,
                              void* d_out, int out_size, void* d_ws, size_t ws_size,
                              hipStream_t stream)
{
    (void)in_sizes; (void)n_in; (void)out_size; (void)ws_size;
    const float* x   = (const float*)d_in[0];
    const float* Fvv = (const float*)d_in[1];
    const float* Fhh = (const float*)d_in[2];
    const float* w1[4] = { (const float*)d_in[3],  (const float*)d_in[7],
                           (const float*)d_in[11], (const float*)d_in[15] };
    const float* b1[4] = { (const float*)d_in[4],  (const float*)d_in[8],
                           (const float*)d_in[12], (const float*)d_in[16] };
    const float* w2[4] = { (const float*)d_in[5],  (const float*)d_in[9],
                           (const float*)d_in[13], (const float*)d_in[17] };
    const float* b2[4] = { (const float*)d_in[6],  (const float*)d_in[10],
                           (const float*)d_in[14], nullptr };

    const size_t FMAP = (size_t)CHN * NPIX * sizeof(float);   // 589824 B
    char* ws = (char*)d_ws;
    float* hA   = (float*)(ws);
    float* hB   = (float*)(ws + FMAP);
    float* t0   = (float*)(ws + 2 * FMAP);
    float* t1   = (float*)(ws + 3 * FMAP);
    float* Abuf = (float*)(ws + 4 * FMAP);
    int*   nIdx = (int*)  (ws + 4 * FMAP + (size_t)NA * LDA * sizeof(float));
    float* logJ = (float*)(ws + 4 * FMAP + (size_t)NA * LDA * sizeof(float) + 4096);

    const float* hcur = x;       // block 0 input/residual is x itself (2 channels)
    float* hnext = hA;
    for (int b = 0; b < 4; ++b) {
        int cin = (b == 0) ? 2 : CHN;
        float scale = 1.0f / sqrtf((float)(b + 1));
        int cnt = cin * NPIX;
        preact_kernel<<<(cnt + 255) / 256, 256, 0, stream>>>(
            hcur, t0, cnt, scale, (b == 0) ? 1 : 0);
        if (b == 0) {
            conv_wmma_kernel<2><<<144, 128, 0, stream>>>(
                t0, w1[b], b1[b], nullptr, t1, 1, /*gelu=*/1);
        } else {
            conv_wmma_kernel<CHN><<<144, 128, 0, stream>>>(
                t0, w1[b], b1[b], nullptr, t1, 1, /*gelu=*/1);
        }
        conv_wmma_kernel<CHN><<<144, 128, 0, stream>>>(
            t1, w2[b], b2[b], hcur, hnext,
            /*resRepeat=*/(b == 0) ? 32 : 1, /*gelu=*/0);
        hcur = hnext;
        hnext = (hnext == hA) ? hB : hA;
    }

    gather_occ_kernel<<<1, 256, 0, stream>>>(x, nIdx);
    logj_kernel<<<1, 256, 0, stream>>>(hcur, logJ);
    assemble_A_kernel<<<(NA * NA + 255) / 256, 256, 0, stream>>>(
        Fvv, Fhh, hcur, nIdx, Abuf);
    pfaffian_kernel<<<1, 1024, 0, stream>>>(Abuf, logJ, (float*)d_out);
}
